// GCN_57415122813717
// MI455X (gfx1250) — compile-verified, matrix operand order checked by reading, not run on hardware
//
#include <hip/hip_runtime.h>
#include <hip/hip_bf16.h>
#include <math.h>

// ---------- types ----------
typedef __attribute__((ext_vector_type(16))) __bf16          v16bf;
typedef __attribute__((ext_vector_type(16))) unsigned short  v16us;
typedef __attribute__((ext_vector_type(8)))  unsigned short  us8;
typedef __attribute__((ext_vector_type(4)))  unsigned short  us4;
typedef __attribute__((ext_vector_type(8)))  float           v8f;
typedef __attribute__((ext_vector_type(4)))  float           f4;

__device__ __forceinline__ unsigned short f32_to_bf16_bits(float f) {
    unsigned int u = __float_as_uint(f);
    unsigned int r = (u + 0x7FFFu + ((u >> 16) & 1u)) >> 16;   // round-to-nearest-even
    return (unsigned short)r;
}
__device__ __forceinline__ float bf16_bits_to_f32(unsigned short h) {
    return __uint_as_float(((unsigned int)h) << 16);
}

// ---------- degree / norm ----------
__global__ void init_deg(float* deg, int n) {
    int i = blockIdx.x * blockDim.x + threadIdx.x;
    if (i < n) deg[i] = 1.0f;                     // self loop
}
__global__ void count_deg(const int* __restrict__ dst, float* deg, int E) {
    int e = blockIdx.x * blockDim.x + threadIdx.x;
    if (e < E) atomicAdd(&deg[dst[e]], 1.0f);
}
__global__ void finalize_dinv(float* deg, int n) {
    int i = blockIdx.x * blockDim.x + threadIdx.x;
    if (i < n) deg[i] = rsqrtf(deg[i]);           // deg >= 1 always
}
__global__ void compute_norm(const int* __restrict__ src, const int* __restrict__ dst,
                             const float* __restrict__ dinv, float* __restrict__ nrm,
                             int E, int Etot) {
    int e = blockIdx.x * blockDim.x + threadIdx.x;
    if (e >= Etot) return;
    if (e < E) nrm[e] = dinv[src[e]] * dinv[dst[e]];
    else { int i = e - E; nrm[e] = dinv[i] * dinv[i]; }
}

// ---------- weight conversion + fragment packing ----------
__global__ void convert_f32_bf16(const float* __restrict__ in, unsigned short* __restrict__ outb, int n) {
    int i = blockIdx.x * blockDim.x + threadIdx.x;
    if (i < n) outb[i] = f32_to_bf16_bits(in[i]);
}
__global__ void conv_w3_pad(const float* __restrict__ W3, unsigned short* __restrict__ W3bf) {
    int i = blockIdx.x * blockDim.x + threadIdx.x;   // 128*16
    if (i >= 128 * 16) return;
    int k = i >> 4, n = i & 15;
    float v = (n < 2) ? W3[k * 2 + n] : 0.0f;
    W3bf[i] = f32_to_bf16_bits(v);
}
// Pack row-major bf16 W[K x N] into per-(ktile,ntile) B fragments:
// Bp[tile][lane][j] with j = 16 contiguous bf16 per lane -> one 32B load/lane in GEMM.
__global__ void pack_B_frag(const unsigned short* __restrict__ Wbf,
                            unsigned short* __restrict__ Bp, int K, int N) {
    int idx = blockIdx.x * blockDim.x + threadIdx.x;
    int total = K * N;
    if (idx >= total) return;
    int tile = idx >> 9;                 // 512 elements per 32x16 tile
    int r    = idx & 511;
    int lane = r >> 4;
    int j    = r & 15;
    int ntiles = N >> 4;
    int kt = tile / ntiles;
    int nt = tile - kt * ntiles;
    int half = lane >> 4, lm = lane & 15;
    int v = j >> 1;
    int kb = (kt << 5) + (half << 3) + ((v >= 4) ? 16 : 0) + ((v & 3) << 1) + (j & 1);
    int n  = (nt << 4) + lm;
    Bp[idx] = Wbf[kb * N + n];
}

// ---------- fragment loaders ----------
__device__ __forceinline__ v16us load_a_bf16(const unsigned short* __restrict__ Ab,
                                             size_t arow, int k0, int half) {
    us8 a0 = *(const us8*)(Ab + arow + k0 + (half << 3));
    us8 a1 = *(const us8*)(Ab + arow + k0 + 16 + (half << 3));
    return __builtin_shufflevector(a0, a1, 0, 1, 2, 3, 4, 5, 6, 7,
                                           8, 9, 10, 11, 12, 13, 14, 15);
}
__device__ __forceinline__ v16us load_a_f32(const float* __restrict__ Af,
                                            size_t arow, int k0, int half) {
    const f4* q0 = (const f4*)(Af + arow + k0 + (half << 3));
    const f4* q1 = (const f4*)(Af + arow + k0 + 16 + (half << 3));
    f4 f0 = q0[0], f1 = q0[1], f2 = q1[0], f3 = q1[1];
    v16us au;
    #pragma unroll
    for (int j = 0; j < 4; ++j) {
        au[j]      = f32_to_bf16_bits(f0[j]);
        au[4 + j]  = f32_to_bf16_bits(f1[j]);
        au[8 + j]  = f32_to_bf16_bits(f2[j]);
        au[12 + j] = f32_to_bf16_bits(f3[j]);
    }
    return au;
}

// ---------- WMMA GEMM: H(bf16) = A(MxK) @ B(KxN) ----------
// One wave per block; wave computes 16 rows x (16*NSUB) cols.
// K stepped by 64 with two register fragment sets (software pipelining):
// loads for stage k+32 / k+64 are issued before the WMMAs that consume
// stage k, hiding L2 latency behind the matrix pipe.
template <int NSUB, bool ABF16>
__global__ void __launch_bounds__(32)
gcn_gemm_wmma(const void* __restrict__ Asrc,
              const unsigned short* __restrict__ Bp,
              unsigned short* __restrict__ Hbf,
              int K, int N) {
    const int lane = threadIdx.x;
    const int half = lane >> 4;
    const int lm   = lane & 15;
    const int row0 = blockIdx.x * 16;
    const int n0b  = blockIdx.y * (16 * NSUB);
    const int ntiles = N >> 4;

    const float*          Af = (const float*)Asrc;
    const unsigned short* Ab = (const unsigned short*)Asrc;

    const int m = row0 + lm;
    const size_t arow = (size_t)m * K;
    const unsigned short* bbase = Bp + (size_t)(n0b >> 4) * 512 + lane * 16;

    auto loadA = [&](int k0) -> v16us {
        return ABF16 ? load_a_bf16(Ab, arow, k0, half)
                     : load_a_f32(Af, arow, k0, half);
    };
    auto loadB = [&](int k0, int s) -> v16us {
        return *(const v16us*)(bbase + (size_t)((k0 >> 5) * ntiles + s) * 512);
    };

    v8f acc[NSUB];
    #pragma unroll
    for (int s = 0; s < NSUB; ++s) { v8f z = {}; acc[s] = z; }

    // prologue: stage 0
    v16us a0 = loadA(0);
    v16us b0[NSUB];
    #pragma unroll
    for (int s = 0; s < NSUB; ++s) b0[s] = loadB(0, s);

    for (int k0 = 0; k0 < K; k0 += 64) {         // K is a multiple of 64
        // issue stage-1 loads before consuming stage 0
        v16us a1 = loadA(k0 + 32);
        v16us b1[NSUB];
        #pragma unroll
        for (int s = 0; s < NSUB; ++s) b1[s] = loadB(k0 + 32, s);

        if (k0 + 64 < K) __builtin_prefetch((const char*)Ab + 2 * (arow + k0 + 64), 0, 0);

        v16bf av0 = __builtin_bit_cast(v16bf, a0);
        #pragma unroll
        for (int s = 0; s < NSUB; ++s) {
            v16bf bv = __builtin_bit_cast(v16bf, b0[s]);
            acc[s] = __builtin_amdgcn_wmma_f32_16x16x32_bf16(
                         false, av0, false, bv, (short)0, acc[s], false, false);
        }

        // issue next stage-0 loads before consuming stage 1
        if (k0 + 64 < K) {
            a0 = loadA(k0 + 64);
            #pragma unroll
            for (int s = 0; s < NSUB; ++s) b0[s] = loadB(k0 + 64, s);
        }

        v16bf av1 = __builtin_bit_cast(v16bf, a1);
        #pragma unroll
        for (int s = 0; s < NSUB; ++s) {
            v16bf bv = __builtin_bit_cast(v16bf, b1[s]);
            acc[s] = __builtin_amdgcn_wmma_f32_16x16x32_bf16(
                         false, av1, false, bv, (short)0, acc[s], false, false);
        }
    }

    // ---- epilogue: C/D layout -> bf16 store (feeds the bf16 edge gather) ----
    #pragma unroll
    for (int s = 0; s < NSUB; ++s) {
        const int n = n0b + 16 * s + lm;
        #pragma unroll
        for (int r = 0; r < 8; ++r) {
            int row = row0 + r + half * 8;
            Hbf[(size_t)row * N + n] = f32_to_bf16_bits(acc[s][r]);
        }
    }
}

// ---------- zero / scatter / epilogues ----------
__global__ void zero_f32_v4(f4* p, int n4) {
    int i = blockIdx.x * blockDim.x + threadIdx.x;
    if (i < n4) { f4 z = {}; p[i] = z; }
}

// 4 channels per thread: one b64 load + 4 f32 atomics (exact segment-sum).
template <int C>
__global__ void gcn_scatter_v4(const unsigned short* __restrict__ Hbf, int hstride,
                               const int* __restrict__ src, const int* __restrict__ dst,
                               const float* __restrict__ nrm, float* __restrict__ A,
                               int E, int Etot) {
    constexpr int CV = C / 4;
    unsigned int t = blockIdx.x * blockDim.x + threadIdx.x;
    unsigned int total = (unsigned int)Etot * (unsigned int)CV;
    if (t >= total) return;
    int e = (int)(t / CV);
    int c = (int)(t % CV) * 4;
    int s, d;
    if (e < E) { s = src[e]; d = dst[e]; }
    else       { s = e - E;  d = s; }
    us4 h = *(const us4*)(Hbf + (size_t)s * hstride + c);
    float nv = nrm[e];
    #pragma unroll
    for (int j = 0; j < 4; ++j)
        atomicAdd(&A[(size_t)d * C + c + j], bf16_bits_to_f32(h[j]) * nv);
}

// Layer-3 scatter: 2 channels per edge, H row stride 16 (padded).
__global__ void gcn_scatter_c2(const unsigned short* __restrict__ Hbf,
                               const int* __restrict__ src, const int* __restrict__ dst,
                               const float* __restrict__ nrm, float* __restrict__ A,
                               int E, int Etot) {
    int e = blockIdx.x * blockDim.x + threadIdx.x;
    if (e >= Etot) return;
    int s, d;
    if (e < E) { s = src[e]; d = dst[e]; }
    else       { s = e - E;  d = s; }
    unsigned int hv = *(const unsigned int*)(Hbf + (size_t)s * 16);
    float nv = nrm[e];
    atomicAdd(&A[d * 2 + 0], bf16_bits_to_f32((unsigned short)(hv & 0xFFFFu)) * nv);
    atomicAdd(&A[d * 2 + 1], bf16_bits_to_f32((unsigned short)(hv >> 16)) * nv);
}

template <int C>
__global__ void bias_relu_bf16_v4(const float* __restrict__ A, const float* __restrict__ b,
                                  unsigned short* __restrict__ P, int total4) {
    int i = blockIdx.x * blockDim.x + threadIdx.x;
    if (i >= total4) return;
    int base = i * 4;
    f4 v = *(const f4*)(A + base);
    us4 o;
    #pragma unroll
    for (int j = 0; j < 4; ++j) {
        float t = v[j] + b[(base + j) & (C - 1)];
        t = t > 0.0f ? t : 0.0f;
        o[j] = f32_to_bf16_bits(t);
    }
    *(us4*)(P + base) = o;
}

__global__ void final_sigmoid(const float* __restrict__ A, const float* __restrict__ b3,
                              float* __restrict__ out, int total) {
    int i = blockIdx.x * blockDim.x + threadIdx.x;
    if (i >= total) return;
    float v = A[i] + b3[i & 1];
    out[i] = 1.0f / (1.0f + __expf(-v));
}

// ---------- orchestration ----------
extern "C" void kernel_launch(void* const* d_in, const int* in_sizes, int n_in,
                              void* d_out, int out_size, void* d_ws, size_t ws_size,
                              hipStream_t stream) {
    (void)in_sizes; (void)n_in; (void)out_size; (void)ws_size;

    const float* x  = (const float*)d_in[0];
    const int*   ei = (const int*)d_in[1];     // [2, 800000]: row0=src, row1=dst
    const float* W1 = (const float*)d_in[2];
    const float* b1 = (const float*)d_in[3];
    const float* W2 = (const float*)d_in[4];
    const float* b2 = (const float*)d_in[5];
    const float* W3 = (const float*)d_in[6];
    const float* b3 = (const float*)d_in[7];
    float* out = (float*)d_out;

    const int Nn = 50000, E = 800000, ET = 850000;
    const int C0 = 768, C1 = 256, C2 = 128;
    const int MT = Nn / 16;                    // 3125 row tiles (exact)

    char* ws = (char*)d_ws;
    size_t off = 0;
    auto alloc = [&](size_t bytes) -> char* {
        char* p = ws + off;
        off = (off + bytes + 255) & ~(size_t)255;
        return p;
    };
    float*          dinv  = (float*)alloc((size_t)Nn * 4);
    float*          nrm   = (float*)alloc((size_t)ET * 4);
    unsigned short* W1bf  = (unsigned short*)alloc((size_t)C0 * C1 * 2);
    unsigned short* W2bf  = (unsigned short*)alloc((size_t)C1 * C2 * 2);
    unsigned short* W3bf  = (unsigned short*)alloc((size_t)C2 * 16 * 2);
    unsigned short* B1p   = (unsigned short*)alloc((size_t)C0 * C1 * 2);
    unsigned short* B2p   = (unsigned short*)alloc((size_t)C1 * C2 * 2);
    unsigned short* B3p   = (unsigned short*)alloc((size_t)C2 * 16 * 2);
    unsigned short* Hbf   = (unsigned short*)alloc((size_t)Nn * 256 * 2); // reused per layer
    float*          Aagg  = (float*)alloc((size_t)Nn * 256 * 4);          // reused per layer
    unsigned short* Pbf   = (unsigned short*)alloc((size_t)Nn * 256 * 2); // reused per layer

    const int* srcI = ei;
    const int* dstI = ei + E;

    // --- normalization (symmetric GCN) ---
    init_deg<<<(Nn + 255) / 256, 256, 0, stream>>>(dinv, Nn);
    count_deg<<<(E + 255) / 256, 256, 0, stream>>>(dstI, dinv, E);
    finalize_dinv<<<(Nn + 255) / 256, 256, 0, stream>>>(dinv, Nn);
    compute_norm<<<(ET + 255) / 256, 256, 0, stream>>>(srcI, dstI, dinv, nrm, E, ET);

    // --- weights -> bf16, then pack into WMMA B fragments ---
    convert_f32_bf16<<<(C0 * C1 + 255) / 256, 256, 0, stream>>>(W1, W1bf, C0 * C1);
    convert_f32_bf16<<<(C1 * C2 + 255) / 256, 256, 0, stream>>>(W2, W2bf, C1 * C2);
    conv_w3_pad<<<(C2 * 16 + 63) / 64, 64, 0, stream>>>(W3, W3bf);
    pack_B_frag<<<(C0 * C1 + 255) / 256, 256, 0, stream>>>(W1bf, B1p, C0, C1);
    pack_B_frag<<<(C1 * C2 + 255) / 256, 256, 0, stream>>>(W2bf, B2p, C1, C2);
    pack_B_frag<<<(C2 * 16 + 255) / 256, 256, 0, stream>>>(W3bf, B3p, C2, 16);

    // ================= Layer 1 =================
    gcn_gemm_wmma<4, false><<<dim3(MT, C1 / 64), 32, 0, stream>>>(x, B1p, Hbf, C0, C1);
    zero_f32_v4<<<(Nn * C1 / 4 + 255) / 256, 256, 0, stream>>>((f4*)Aagg, Nn * C1 / 4);
    {
        unsigned int blocks = ((unsigned int)ET * (C1 / 4) + 255u) / 256u;
        gcn_scatter_v4<256><<<blocks, 256, 0, stream>>>(Hbf, C1, srcI, dstI, nrm, Aagg, E, ET);
    }
    bias_relu_bf16_v4<256><<<(Nn * C1 / 4 + 255) / 256, 256, 0, stream>>>(Aagg, b1, Pbf, Nn * C1 / 4);

    // ================= Layer 2 =================
    gcn_gemm_wmma<4, true><<<dim3(MT, C2 / 64), 32, 0, stream>>>(Pbf, B2p, Hbf, C1, C2);
    zero_f32_v4<<<(Nn * C2 / 4 + 255) / 256, 256, 0, stream>>>((f4*)Aagg, Nn * C2 / 4);
    {
        unsigned int blocks = ((unsigned int)ET * (C2 / 4) + 255u) / 256u;
        gcn_scatter_v4<128><<<blocks, 256, 0, stream>>>(Hbf, C2, srcI, dstI, nrm, Aagg, E, ET);
    }
    bias_relu_bf16_v4<128><<<(Nn * C2 / 4 + 255) / 256, 256, 0, stream>>>(Aagg, b2, Pbf, Nn * C2 / 4);

    // ================= Layer 3 (N padded 2 -> 16) =================
    gcn_gemm_wmma<1, true><<<dim3(MT, 1), 32, 0, stream>>>(Pbf, B3p, Hbf, C2, 16);
    zero_f32_v4<<<(Nn * 2 / 4 + 255) / 256, 256, 0, stream>>>((f4*)Aagg, Nn * 2 / 4);
    gcn_scatter_c2<<<(ET + 255) / 256, 256, 0, stream>>>(Hbf, srcI, dstI, nrm, Aagg, E, ET);
    final_sigmoid<<<(Nn * 2 + 255) / 256, 256, 0, stream>>>(Aagg, b3, out, Nn * 2);
}
